// DecoderBlock_2113123910089
// MI455X (gfx1250) — matrix operand order repaired and checked by reading, not compiled, not run on hardware
//
#include <hip/hip_runtime.h>
#include <hip/hip_bf16.h>
#include <stdint.h>

#define N_EMBD 768
#define N_HEADS 12
#define HDIM 64
#define BB 2
#define TT 2048
#define MROWS (BB*TT)   // 4096

typedef __attribute__((ext_vector_type(16))) __bf16 v16bf;
typedef __attribute__((ext_vector_type(8)))  __bf16 v8bf;
typedef __attribute__((ext_vector_type(4)))  __bf16 v4bf;
typedef __attribute__((ext_vector_type(8)))  float  v8f;

union Frag16 {
  v16bf bf;
  uint4 u4[2];
  unsigned short us[16];
};

// Native bf16 conversion -> v_cvt(_pk)_bf16_f32
__device__ __forceinline__ unsigned short f2bf(float f) {
  union { __bf16 h; unsigned short u; } t;
  t.h = (__bf16)f;
  return t.u;
}

__device__ __forceinline__ v8f wmma_bf16(const Frag16& a, const Frag16& b, v8f c) {
  return __builtin_amdgcn_wmma_f32_16x16x32_bf16(false, a.bf, false, b.bf,
                                                 (short)0, c, false, false);
}

// LDS byte offset of a __shared__ pointer (generic low 32 bits = wave-relative
// LDS offset on amdgcn: shared aperture lives in addr[63:32]).
__device__ __forceinline__ unsigned lds_off(const void* p) {
  return (unsigned)(unsigned long long)p;
}

// ---------------------------------------------------------------------------
// Weight pre-pass: WT[n][k] = bf16(W[k][n])  (transpose + convert, 32x32 tiles)
// ---------------------------------------------------------------------------
__global__ __launch_bounds__(256) void cvt_transpose_w_kernel(
    const float* __restrict__ W, unsigned short* __restrict__ WT, int K, int N)
{
  __shared__ float tile[32][33];
  const int k0 = blockIdx.y * 32, n0 = blockIdx.x * 32;
  const int tx = threadIdx.x & 31, ty = threadIdx.x >> 5;   // 32x8
  #pragma unroll
  for (int i = 0; i < 32; i += 8)
    tile[ty + i][tx] = W[(size_t)(k0 + ty + i) * N + n0 + tx];
  __syncthreads();
  #pragma unroll
  for (int i = 0; i < 32; i += 8)
    WT[(size_t)(n0 + ty + i) * K + k0 + tx] = f2bf(tile[tx][ty + i]);
}

// ---------------------------------------------------------------------------
// GEMM: C[M,N] = act(A[M,K] @ W[K,N] + bias[N]).
// A: fp32 row-major, converted to bf16 while staging (vector b128 path).
// W: bf16 PRE-TRANSPOSED [N][K], staged by global_load_async_to_lds_b128.
// DOUBLE-BUFFERED LDS: tile i+1's async DMA + A-pack run concurrently with
// tile i's WMMAs; one s_wait_asynccnt + one barrier per iteration.
// Block = 256 threads = 8 waves in 2(M)x4(N); block tile 64x128; K step 32.
// ---------------------------------------------------------------------------
__global__ __launch_bounds__(256) void gemm_bf16_kernel(
    const float* __restrict__ A, const unsigned short* __restrict__ WT,
    const float* __restrict__ bias, float* __restrict__ C,
    int M, int N, int K, int relu)
{
  __shared__ __align__(16) unsigned short sA[2][64*32];    // [buf][m][k]
  __shared__ __align__(16) unsigned short sB[2][128*32];   // [buf][n][k]
  const int tid = threadIdx.x;
  const int bm  = blockIdx.y * 64;
  const int bn  = blockIdx.x * 128;
  const int wave = tid >> 5;
  const int lane = tid & 31;
  const int wm = wave >> 2;        // 0..1
  const int wn = wave & 3;         // 0..3
  const int hl  = lane >> 4;       // lane half
  const int l16 = lane & 15;
  const int arow = tid >> 2;       // 0..63
  const int acol = (tid & 3) * 8;  // 0,8,16,24

  v8f acc[2][2] = {};

  auto stage = [&](int kt, int buf) {
    // W tile: async DMA 8KB global->LDS, 512 x b128 chunks, 2 per thread
    #pragma unroll
    for (int i = 0; i < 2; ++i) {
      int c = tid + i * 256;                 // 0..511
      int n = c >> 2, kc = (c & 3) * 8;
      unsigned long long gp =
          (unsigned long long)(WT + (size_t)(bn + n) * K + kt + kc);
      unsigned lo = lds_off(&sB[buf][n * 32 + kc]);
      asm volatile("global_load_async_to_lds_b128 %0, %1, off"
                   :: "v"(lo), "v"(gp) : "memory");
    }
    // A tile: 8 floats/thread via 2x b128 loads -> 8-wide bf16 vector
    // (cvt_pk results feed the ds_store_b128 directly, no repacking)
    const float4* ap = (const float4*)&A[(size_t)(bm + arow) * K + kt + acol];
    float4 f0 = ap[0], f1 = ap[1];
    v8bf pk;
    pk[0] = (__bf16)f0.x; pk[1] = (__bf16)f0.y;
    pk[2] = (__bf16)f0.z; pk[3] = (__bf16)f0.w;
    pk[4] = (__bf16)f1.x; pk[5] = (__bf16)f1.y;
    pk[6] = (__bf16)f1.z; pk[7] = (__bf16)f1.w;
    *(v8bf*)&sA[buf][arow * 32 + acol] = pk;
  };

  // prologue: stage tile 0
  stage(0, 0);
  asm volatile("s_wait_asynccnt 0x0" ::: "memory");
  __syncthreads();

  for (int kt = 0; kt < K; kt += 32) {
    const int cur = (kt >> 5) & 1;
    // overlap: stage tile i+1 into the other buffer while computing tile i
    if (kt + 32 < K) stage(kt + 32, cur ^ 1);

    // A-fragment 16x32: lanes 0-15 M=lane K{0..7,16..23}; lanes 16-31 K{8..15,24..31}
    Frag16 fa[2], fb[2];
    #pragma unroll
    for (int mt = 0; mt < 2; ++mt) {
      const unsigned short* base = &sA[cur][(wm*32 + mt*16 + l16) * 32];
      fa[mt].u4[0] = *(const uint4*)(base + hl*8);
      fa[mt].u4[1] = *(const uint4*)(base + 16 + hl*8);
    }
    // B-fragment 32x16: lane = N, lanes 0-15 K{0..7,16..23}; lanes 16-31 K{8..15,24..31}
    #pragma unroll
    for (int nt = 0; nt < 2; ++nt) {
      const unsigned short* base = &sB[cur][(wn*32 + nt*16 + l16) * 32];
      fb[nt].u4[0] = *(const uint4*)(base + hl*8);
      fb[nt].u4[1] = *(const uint4*)(base + 16 + hl*8);
    }
    #pragma unroll
    for (int mt = 0; mt < 2; ++mt)
      #pragma unroll
      for (int nt = 0; nt < 2; ++nt)
        acc[mt][nt] = wmma_bf16(fa[mt], fb[nt], acc[mt][nt]);

    asm volatile("s_wait_asynccnt 0x0" ::: "memory");
    __syncthreads();
  }

  #pragma unroll
  for (int mt = 0; mt < 2; ++mt)
    #pragma unroll
    for (int nt = 0; nt < 2; ++nt)
      #pragma unroll
      for (int r = 0; r < 8; ++r) {
        int row = bm + wm*32 + mt*16 + r + hl*8;
        int col = bn + wn*32 + nt*16 + l16;
        float v = acc[mt][nt][r] + bias[col];
        if (relu) v = fmaxf(v, 0.0f);
        C[(size_t)row*N + col] = v;
      }
}

// ---------------------------------------------------------------------------
// Pack fp32 projections into bf16 head-major buffers (4 channels / thread):
//   Qb,Kb: [b,h,t,d]   Vb: [b,h,d,t] (transposed for P@V B-fragments)
// ---------------------------------------------------------------------------
__global__ __launch_bounds__(256) void pack_qkv_kernel(
    const float* __restrict__ qs, int qstride,
    const float* __restrict__ ks, int kstride,
    const float* __restrict__ vs, int vstride,
    unsigned short* __restrict__ Qb,
    unsigned short* __restrict__ Kb,
    unsigned short* __restrict__ Vb)
{
  size_t i = (size_t)blockIdx.x * 256 + threadIdx.x;
  const size_t total = (size_t)BB * TT * (N_EMBD / 4);
  if (i >= total) return;
  int c = (int)(i % (N_EMBD / 4)) * 4;
  int t = (int)((i / (N_EMBD / 4)) % TT);
  int b = (int)(i / ((size_t)(N_EMBD / 4) * TT));
  int h = c >> 6;
  int d = c & 63;
  size_t bh  = (size_t)(b * N_HEADS + h);
  size_t row = (size_t)b * TT + t;

  float4 q = *(const float4*)&qs[row * (size_t)qstride + c];
  float4 k = *(const float4*)&ks[row * (size_t)kstride + c];
  float4 v = *(const float4*)&vs[row * (size_t)vstride + c];

  v4bf qp; qp[0] = (__bf16)q.x; qp[1] = (__bf16)q.y;
           qp[2] = (__bf16)q.z; qp[3] = (__bf16)q.w;
  v4bf kp; kp[0] = (__bf16)k.x; kp[1] = (__bf16)k.y;
           kp[2] = (__bf16)k.z; kp[3] = (__bf16)k.w;
  *(v4bf*)&Qb[(bh*TT + t)*HDIM + d] = qp;
  *(v4bf*)&Kb[(bh*TT + t)*HDIM + d] = kp;

  unsigned short* vd = &Vb[((bh*HDIM) + d)*TT + t];
  vd[0*TT] = f2bf(v.x);
  vd[1*TT] = f2bf(v.y);
  vd[2*TT] = f2bf(v.z);
  vd[3*TT] = f2bf(v.w);
}

// ---------------------------------------------------------------------------
// Flash attention (WMMA): block = 128 threads = 4 waves, each wave owns 16
// query rows of one (b,h); online softmax; key blocks of 32.
// ---------------------------------------------------------------------------
__global__ __launch_bounds__(128) void flash_attn_kernel(
    const unsigned short* __restrict__ Qb,
    const unsigned short* __restrict__ Kb,
    const unsigned short* __restrict__ Vb,
    float* __restrict__ Y, int causal)
{
  __shared__ __align__(16) unsigned short spb[4][16][32];   // per-wave bf16 P
  const int lane = threadIdx.x & 31;
  const int w    = threadIdx.x >> 5;
  const int hl   = lane >> 4;
  const int l16  = lane & 15;
  const int bh = blockIdx.y;
  const int b  = bh / N_HEADS;
  const int h  = bh % N_HEADS;
  const int q0 = blockIdx.x * 64 + w * 16;

  const unsigned short* Q = Qb + ((size_t)bh*TT + q0) * HDIM;
  const unsigned short* K = Kb + (size_t)bh*TT*HDIM;
  const unsigned short* V = Vb + (size_t)bh*HDIM*TT;

  // Q as A-fragments (16 rows x 64 dims = 2 fragments of K=32)
  Frag16 fq[2];
  #pragma unroll
  for (int kb = 0; kb < 2; ++kb) {
    const unsigned short* r = Q + l16*HDIM + kb*32;
    fq[kb].u4[0] = *(const uint4*)(r + hl*8);
    fq[kb].u4[1] = *(const uint4*)(r + 16 + hl*8);
  }

  float mi[8], li[8];
  #pragma unroll
  for (int r = 0; r < 8; ++r) { mi[r] = -1e30f; li[r] = 0.0f; }
  v8f o[4] = {};

  const int jend = causal ? ((q0 + 16 + 31) & ~31) : TT;
  const float scale = 0.125f;   // 1/sqrt(64)

  for (int j = 0; j < jend; j += 32) {
    // K^T as B-fragments: [t = key tile 0/1][kb = dim half 0/1]
    Frag16 fk[2][2];
    #pragma unroll
    for (int t = 0; t < 2; ++t)
      #pragma unroll
      for (int kb = 0; kb < 2; ++kb) {
        const unsigned short* base = K + (size_t)(j + t*16 + l16)*HDIM + kb*32 + hl*8;
        fk[t][kb].u4[0] = *(const uint4*)(base);
        fk[t][kb].u4[1] = *(const uint4*)(base + 16);
      }
    v8f s0 = {}, s1 = {};
    s0 = wmma_bf16(fq[0], fk[0][0], s0);
    s0 = wmma_bf16(fq[1], fk[0][1], s0);
    s1 = wmma_bf16(fq[0], fk[1][0], s1);
    s1 = wmma_bf16(fq[1], fk[1][1], s1);

    // scale, mask, online-softmax row stats (row = VGPR index, col = lane%16)
    #pragma unroll
    for (int r = 0; r < 8; ++r) {
      int m  = r + hl*8;
      int qi = q0 + m;
      float a0 = s0[r]*scale;
      float a1 = s1[r]*scale;
      if (causal) {
        if (j + l16 > qi)      a0 = -1e30f;
        if (j + 16 + l16 > qi) a1 = -1e30f;
      }
      float mx = fmaxf(a0, a1);
      #pragma unroll
      for (int off = 1; off < 16; off <<= 1) mx = fmaxf(mx, __shfl_xor(mx, off, 32));
      float mnew = fmaxf(mi[r], mx);
      float corr = __expf(mi[r] - mnew);
      float p0 = __expf(a0 - mnew);
      float p1 = __expf(a1 - mnew);
      float ls = p0 + p1;
      #pragma unroll
      for (int off = 1; off < 16; off <<= 1) ls += __shfl_xor(ls, off, 32);
      li[r] = li[r]*corr + ls;
      mi[r] = mnew;
      o[0][r] *= corr; o[1][r] *= corr; o[2][r] *= corr; o[3][r] *= corr;
      spb[w][m][l16]      = f2bf(p0);
      spb[w][m][16 + l16] = f2bf(p1);
    }

    // same-wave LDS RAW: LDS in-order per wave; ensure stores landed
    asm volatile("s_wait_dscnt 0x0" ::: "memory");

    // P as A-fragment (16 q rows x 32 keys) read straight back as b128s
    Frag16 fp;
    fp.u4[0] = *(const uint4*)&spb[w][l16][hl*8];
    fp.u4[1] = *(const uint4*)&spb[w][l16][16 + hl*8];

    // O += P @ V  (V transposed [d][t] -> contiguous B-fragment loads)
    #pragma unroll
    for (int n16 = 0; n16 < 4; ++n16) {
      Frag16 fv;
      const unsigned short* vb = V + (size_t)(n16*16 + l16)*TT + j + hl*8;
      fv.u4[0] = *(const uint4*)(vb);
      fv.u4[1] = *(const uint4*)(vb + 16);
      o[n16] = wmma_bf16(fp, fv, o[n16]);
    }
  }

  #pragma unroll
  for (int r = 0; r < 8; ++r) {
    int m = r + hl*8;
    float inv = 1.0f / li[r];
    size_t rowbase = ((size_t)b*TT + q0 + m)*N_EMBD + h*HDIM + l16;
    #pragma unroll
    for (int n16 = 0; n16 < 4; ++n16)
      Y[rowbase + n16*16] = o[n16][r] * inv;
  }
}

// ---------------------------------------------------------------------------
// out = LayerNorm(a + b) * g + beta  (one block per 768-wide row)
// ---------------------------------------------------------------------------
__global__ __launch_bounds__(256) void ln_add_kernel(
    const float* __restrict__ a, const float* __restrict__ b,
    const float* __restrict__ g, const float* __restrict__ beta,
    float* __restrict__ out)
{
  const int row = blockIdx.x;
  const int tid = threadIdx.x;
  const size_t base = (size_t)row * N_EMBD;
  float v[3];
  float s = 0.f, s2 = 0.f;
  #pragma unroll
  for (int i = 0; i < 3; ++i) {
    int idx = tid + i*256;
    float x = a[base+idx] + b[base+idx];
    v[i] = x; s += x; s2 += x*x;
  }
  #pragma unroll
  for (int off = 1; off < 32; off <<= 1) {
    s  += __shfl_xor(s,  off, 32);
    s2 += __shfl_xor(s2, off, 32);
  }
  __shared__ float ssum[8], ssq[8];
  if ((tid & 31) == 0) { ssum[tid>>5] = s; ssq[tid>>5] = s2; }
  __syncthreads();
  float ts = 0.f, tq = 0.f;
  #pragma unroll
  for (int i = 0; i < 8; ++i) { ts += ssum[i]; tq += ssq[i]; }
  const float inv = 1.0f / (float)N_EMBD;
  float mu  = ts * inv;
  float var = tq * inv - mu*mu;
  float rstd = rsqrtf(var + 1e-5f);
  #pragma unroll
  for (int i = 0; i < 3; ++i) {
    int idx = tid + i*256;
    out[base+idx] = (v[i]-mu)*rstd*g[idx] + beta[idx];
  }
}

// ---------------------------------------------------------------------------
extern "C" void kernel_launch(void* const* d_in, const int* in_sizes, int n_in,
                              void* d_out, int out_size, void* d_ws, size_t ws_size,
                              hipStream_t stream)
{
  const float* x    = (const float*)d_in[0];
  const float* enc  = (const float*)d_in[1];
  const float* ln1g = (const float*)d_in[2];
  const float* ln1b = (const float*)d_in[3];
  const float* qkvw = (const float*)d_in[4];
  const float* qkvb = (const float*)d_in[5];
  const float* projw= (const float*)d_in[6];
  const float* projb= (const float*)d_in[7];
  const float* ln2g = (const float*)d_in[8];
  const float* ln2b = (const float*)d_in[9];
  const float* caqw = (const float*)d_in[10];
  const float* caqb = (const float*)d_in[11];
  const float* cakw = (const float*)d_in[12];
  const float* cakb = (const float*)d_in[13];
  const float* cavw = (const float*)d_in[14];
  const float* cavb = (const float*)d_in[15];
  const float* caow = (const float*)d_in[16];
  const float* caob = (const float*)d_in[17];
  const float* ln3g = (const float*)d_in[18];
  const float* ln3b = (const float*)d_in[19];
  const float* fcw  = (const float*)d_in[20];
  const float* fcb  = (const float*)d_in[21];
  const float* fc2w = (const float*)d_in[22];
  const float* fc2b = (const float*)d_in[23];
  float* out = (float*)d_out;

  char* ws = (char*)d_ws;
  const size_t SZ_BTC_F = (size_t)MROWS * N_EMBD;                 // floats
  const size_t SZ_BTC   = SZ_BTC_F * sizeof(float);
  const size_t SZ_BIG   = (size_t)MROWS * 3072 * sizeof(float);
  const size_t SZ_HB    = (size_t)MROWS * N_EMBD * sizeof(unsigned short);

  float* BIG = (float*)ws;                                         // qkv / ck,cv / mlp hidden
  unsigned short* Qb = (unsigned short*)(ws + SZ_BIG);
  unsigned short* Kb = (unsigned short*)(ws + SZ_BIG + SZ_HB);
  unsigned short* Vb = (unsigned short*)(ws + SZ_BIG + 2*SZ_HB);
  float* Y  = (float*)(ws + SZ_BIG + 3*SZ_HB);
  float* T1 = Y  + SZ_BTC_F;
  float* X1 = T1 + SZ_BTC_F;
  float* X2 = X1 + SZ_BTC_F;

  // bf16 transposed weight region
  size_t woff = SZ_BIG + 3*SZ_HB + 4*SZ_BTC;
  unsigned short* WTqkv = (unsigned short*)(ws + woff); woff += (size_t)768*2304*2;
  unsigned short* WTproj= (unsigned short*)(ws + woff); woff += (size_t)768*768*2;
  unsigned short* WTcaq = (unsigned short*)(ws + woff); woff += (size_t)768*768*2;
  unsigned short* WTcak = (unsigned short*)(ws + woff); woff += (size_t)768*768*2;
  unsigned short* WTcav = (unsigned short*)(ws + woff); woff += (size_t)768*768*2;
  unsigned short* WTcao = (unsigned short*)(ws + woff); woff += (size_t)768*768*2;
  unsigned short* WTfc  = (unsigned short*)(ws + woff); woff += (size_t)768*3072*2;
  unsigned short* WTfc2 = (unsigned short*)(ws + woff); woff += (size_t)3072*768*2;

  dim3 blk(256);
  const int packBlocks = (int)(((size_t)BB*TT*(N_EMBD/4) + 255) / 256);
  dim3 attnGrid(TT/64, BB*N_HEADS);

  // 0) weight convert + transpose (fp32 [K][N] -> bf16 [N][K])
  cvt_transpose_w_kernel<<<dim3(2304/32,  768/32), blk, 0, stream>>>(qkvw, WTqkv,  768, 2304);
  cvt_transpose_w_kernel<<<dim3( 768/32,  768/32), blk, 0, stream>>>(projw, WTproj, 768,  768);
  cvt_transpose_w_kernel<<<dim3( 768/32,  768/32), blk, 0, stream>>>(caqw, WTcaq,  768,  768);
  cvt_transpose_w_kernel<<<dim3( 768/32,  768/32), blk, 0, stream>>>(cakw, WTcak,  768,  768);
  cvt_transpose_w_kernel<<<dim3( 768/32,  768/32), blk, 0, stream>>>(cavw, WTcav,  768,  768);
  cvt_transpose_w_kernel<<<dim3( 768/32,  768/32), blk, 0, stream>>>(caow, WTcao,  768,  768);
  cvt_transpose_w_kernel<<<dim3(3072/32,  768/32), blk, 0, stream>>>(fcw,  WTfc,   768, 3072);
  cvt_transpose_w_kernel<<<dim3( 768/32, 3072/32), blk, 0, stream>>>(fc2w, WTfc2, 3072,  768);

  // 1) qkv = x @ qkvw + qkvb                     [4096 x 2304]
  gemm_bf16_kernel<<<dim3(2304/128, MROWS/64), blk, 0, stream>>>(x, WTqkv, qkvb, BIG, MROWS, 2304, 768, 0);
  // 2) pack self-attn Q/K/V to bf16 head-major
  pack_qkv_kernel<<<packBlocks, blk, 0, stream>>>(BIG, 2304, BIG+768, 2304, BIG+1536, 2304, Qb, Kb, Vb);
  // 3) causal flash attention -> Y
  flash_attn_kernel<<<attnGrid, dim3(128), 0, stream>>>(Qb, Kb, Vb, Y, 1);
  // 4) sa = Y @ projw + projb -> T1
  gemm_bf16_kernel<<<dim3(6, MROWS/64), blk, 0, stream>>>(Y, WTproj, projb, T1, MROWS, 768, 768, 0);
  // 5) x1 = LN(sa + x)
  ln_add_kernel<<<MROWS, blk, 0, stream>>>(T1, x, ln1g, ln1b, X1);
  // 6) cq = x1 @ ca_q_w -> Y (reuse)
  gemm_bf16_kernel<<<dim3(6, MROWS/64), blk, 0, stream>>>(X1, WTcaq, caqb, Y, MROWS, 768, 768, 0);
  // 7) ck = enc @ ca_k_w -> BIG
  gemm_bf16_kernel<<<dim3(6, MROWS/64), blk, 0, stream>>>(enc, WTcak, cakb, BIG, MROWS, 768, 768, 0);
  // 8) cv = enc @ ca_v_w -> BIG + BTC
  gemm_bf16_kernel<<<dim3(6, MROWS/64), blk, 0, stream>>>(enc, WTcav, cavb, BIG + SZ_BTC_F, MROWS, 768, 768, 0);
  // 9) pack cross-attn Q/K/V
  pack_qkv_kernel<<<packBlocks, blk, 0, stream>>>(Y, 768, BIG, 768, BIG + SZ_BTC_F, 768, Qb, Kb, Vb);
  // 10) non-causal flash attention -> Y
  flash_attn_kernel<<<attnGrid, dim3(128), 0, stream>>>(Qb, Kb, Vb, Y, 0);
  // 11) ca = Y @ ca_o_w -> T1
  gemm_bf16_kernel<<<dim3(6, MROWS/64), blk, 0, stream>>>(Y, WTcao, caob, T1, MROWS, 768, 768, 0);
  // 12) x2 = LN(ca + x1)
  ln_add_kernel<<<MROWS, blk, 0, stream>>>(T1, X1, ln2g, ln2b, X2);
  // 13) h = relu(x2 @ fc_w + fc_b) -> BIG       [4096 x 3072]
  gemm_bf16_kernel<<<dim3(3072/128, MROWS/64), blk, 0, stream>>>(X2, WTfc, fcb, BIG, MROWS, 3072, 768, 1);
  // 14) h2 = h @ fc2_w + fc2_b -> T1            [K = 3072]
  gemm_bf16_kernel<<<dim3(6, MROWS/64), blk, 0, stream>>>(BIG, WTfc2, fc2b, T1, MROWS, 768, 3072, 0);
  // 15) out = LN(h2 + x2)
  ln_add_kernel<<<MROWS, blk, 0, stream>>>(T1, X2, ln3g, ln3b, out);

  (void)in_sizes; (void)n_in; (void)out_size; (void)ws_size;
}
